// SCAM_17042430230771
// MI455X (gfx1250) — matrix operand order repaired and checked by reading, not compiled, not run on hardware
//
#include <hip/hip_runtime.h>

typedef __attribute__((ext_vector_type(16))) __bf16 v16bf;
typedef __attribute__((ext_vector_type(8)))  __bf16 v8bf;
typedef __attribute__((ext_vector_type(8)))  float  v8f;

#define HW 65536
#define IMGW 256

__device__ __forceinline__ v16bf cat8(v8bf lo, v8bf hi) {
  return __builtin_shufflevector(lo, hi, 0,1,2,3,4,5,6,7,8,9,10,11,12,13,14,15);
}
__device__ __forceinline__ v8f wmma_bf(v16bf a, v16bf b, v8f c) {
  return __builtin_amdgcn_wmma_f32_16x16x32_bf16(false, a, false, b, (short)0, c, false, false);
}

// ---------- elementwise helpers ----------
__global__ __launch_bounds__(256) void k_f2b(const float* __restrict__ s,
                                             __bf16* __restrict__ d, int n) {
  int i = blockIdx.x * 256 + threadIdx.x;
  if (i < n) d[i] = (__bf16)s[i];
}

// OIHW f32 -> GEMM bf16 layout A[co][r*Cin+ci], r = ky*3+kx (taps=9) or identity (taps=1)
__global__ __launch_bounds__(256) void k_prep_w(const float* __restrict__ src,
                                                __bf16* __restrict__ dst,
                                                int Cout, int Cin, int taps) {
  int i = blockIdx.x * 256 + threadIdx.x;
  int total = Cout * Cin * taps;
  if (i >= total) return;
  int r  = i % taps;
  int ci = (i / taps) % Cin;
  int co = i / (taps * Cin);
  dst[(size_t)co * (Cin * taps) + r * Cin + ci] = (__bf16)src[i];
}

// ---------- implicit-GEMM conv (TAPS=9 SAME 3x3, TAPS=1 pointwise) ----------
// Block tile: 64 (co) x 128 (px). 8 waves; wave = 16(co) x 64(px) = 4 WMMA accs.
// Double-buffered LDS, one barrier per K-chunk of 32.
template<int TAPS>
__global__ __launch_bounds__(256) void k_gemm_conv(
    const __bf16* __restrict__ src,   // [Cin][HW] bf16
    const __bf16* __restrict__ Ag,    // [Cout][Cin*TAPS] bf16
    const float*  __restrict__ bias,  // nullable
    __bf16* __restrict__ dstB,        // nullable [Cout][HW]
    float*  __restrict__ dstF,        // nullable [Cout][HW]
    const float* __restrict__ resid,  // nullable [Cout][HW]
    int Cout, int Cin)
{
  __shared__ __bf16 As[2][64 * 40];
  __shared__ __bf16 Bs[2][128 * 40];
  const int tid  = threadIdx.x;
  const int lane = tid & 31;
  const int wave = tid >> 5;
  const int g = lane >> 4;
  const int l15 = lane & 15;
  const int mTile = (wave & 3) * 16;
  const int nBase = (wave >> 2) * 64;
  const int pixBase = blockIdx.x * 128;
  const int h  = pixBase >> 8;
  const int w0 = pixBase & 255;
  const int coBase = blockIdx.y * 64;
  const int K = Cin * TAPS;
  const int cinCh = (Cin + 31) >> 5;
  const int nChunks = cinCh * TAPS;

  // incrementally-maintained load-pointer state (no divisions in loop)
  int cc = 0, rCin = 0;
  int dy = (TAPS == 9) ? -1 : 0;
  int dx = (TAPS == 9) ? -1 : 0;

  const int akk = tid & 31, ai0 = tid >> 5;    // A: 64x32, 8 rows/thread
  const int bn  = tid & 127, bk0 = tid >> 7;   // B: 128x32 (transposed), 16 vals/thread

  __bf16 aR[8];
  __bf16 bR[16];

  auto loadRegs = [&]() {
    int ci0 = cc << 5;
    #pragma unroll
    for (int ii = 0; ii < 8; ++ii) {
      int i  = ai0 + (ii << 3);
      int co = coBase + i, ci = ci0 + akk;
      __bf16 v = (__bf16)0.0f;
      if (co < Cout && ci < Cin) v = Ag[(size_t)co * K + rCin + ci];
      aR[ii] = v;
    }
    int hh = h + dy;
    int ww = w0 + bn + dx;
    bool pix_ok = ((unsigned)hh < 256u) && ((unsigned)ww < 256u);
    int poff = hh * IMGW + ww;
    #pragma unroll
    for (int ii = 0; ii < 16; ++ii) {
      int kk = bk0 + (ii << 1);
      int ci = ci0 + kk;
      __bf16 v = (__bf16)0.0f;
      if (pix_ok && ci < Cin) v = src[(size_t)ci * HW + poff];
      bR[ii] = v;
    }
  };
  auto storeLDS = [&](int b) {
    #pragma unroll
    for (int ii = 0; ii < 8; ++ii) As[b][(ai0 + (ii << 3)) * 40 + akk] = aR[ii];
    #pragma unroll
    for (int ii = 0; ii < 16; ++ii) Bs[b][bn * 40 + bk0 + (ii << 1)] = bR[ii];
  };
  auto advance = [&]() {
    if (++cc == cinCh) {
      cc = 0; rCin += Cin;
      if (TAPS == 9) { if (++dx == 2) { dx = -1; ++dy; } }
    }
  };

  v8f acc[4] = {{}, {}, {}, {}};

  loadRegs(); storeLDS(0); advance();
  __syncthreads();

  for (int ch = 0; ch < nChunks; ++ch) {
    int cur = ch & 1;
    bool more = (ch + 1) < nChunks;
    if (more) loadRegs();                 // global loads overlap WMMA below
    {
      int m = mTile + l15;
      v16bf a = cat8(*(const v8bf*)(&As[cur][m * 40 + g * 8]),
                     *(const v8bf*)(&As[cur][m * 40 + 16 + g * 8]));
      #pragma unroll
      for (int f = 0; f < 4; ++f) {
        int n0 = nBase + f * 16 + l15;
        v16bf bb = cat8(*(const v8bf*)(&Bs[cur][n0 * 40 + g * 8]),
                        *(const v8bf*)(&Bs[cur][n0 * 40 + 16 + g * 8]));
        acc[f] = wmma_bf(a, bb, acc[f]);
      }
    }
    if (more) { storeLDS(cur ^ 1); advance(); }
    __syncthreads();
  }

  // epilogue: C layout lane->(n=lane&15, m=jj+8*(lane>>4))
  int mg = (lane >> 4) * 8;
  #pragma unroll
  for (int jj = 0; jj < 8; ++jj) {
    int co = coBase + mTile + mg + jj;
    if (co >= Cout) continue;
    float bv = bias ? bias[co] : 0.0f;
    size_t row = (size_t)co * HW;
    #pragma unroll
    for (int f = 0; f < 4; ++f) {
      int p = pixBase + nBase + f * 16 + l15;
      float v = acc[f][jj] + bv;
      if (resid) v += resid[row + p];
      if (dstB)  dstB[row + p] = (__bf16)v;
      if (dstF)  dstF[row + p] = v;
    }
  }
}

// ---------- channel attention: attn = (q k^T) * temp, split-K over 8 waves ----------
__global__ __launch_bounds__(256) void k_attn(
    const __bf16* __restrict__ qkv, const float* __restrict__ t1,
    const float* __restrict__ t2, __bf16* __restrict__ attnB)
{
  __shared__ float red[8 * 256];
  int pair = blockIdx.x >> 3;
  int hh   = blockIdx.x & 7;
  int tid = threadIdx.x, lane = tid & 31, wave = tid >> 5;
  int g = lane >> 4;
  int qCh = (pair ? 384 : 256) + hh * 16;
  int kCh = (pair ? 640 : 0)   + hh * 16;
  const __bf16* qrow = qkv + (size_t)(qCh + (lane & 15)) * HW;
  const __bf16* krow = qkv + (size_t)(kCh + (lane & 15)) * HW;
  v8f acc = {};
  for (int c = wave; c < 2048; c += 8) {
    int n0 = c << 5;
    v16bf a = cat8(*(const v8bf*)(qrow + n0 + g * 8),
                   *(const v8bf*)(qrow + n0 + 16 + g * 8));
    v16bf b = cat8(*(const v8bf*)(krow + n0 + g * 8),
                   *(const v8bf*)(krow + n0 + 16 + g * 8));
    acc = wmma_bf(a, b, acc);
  }
  #pragma unroll
  for (int jj = 0; jj < 8; ++jj) red[wave * 256 + jj * 32 + lane] = acc[jj];
  __syncthreads();
  {
    int jj = tid >> 5, l = tid & 31;
    float s = 0.f;
    #pragma unroll
    for (int w = 0; w < 8; ++w) s += red[w * 256 + jj * 32 + l];
    int m = jj + ((l >> 4) << 3);
    int d = l & 15;
    s *= pair ? t2[hh] : t1[hh];
    size_t base = ((size_t)(pair * 8 + hh) * 16 + m) * 32;
    attnB[base + d]      = (__bf16)s;
    attnB[base + 16 + d] = (__bf16)0.0f;   // zero pad K 16..31
  }
}

// ---------- out = attn @ v, written into fusion input slices ----------
__global__ __launch_bounds__(256) void k_av(
    const __bf16* __restrict__ qkv, const __bf16* __restrict__ attnB,
    __bf16* __restrict__ fusedIn)
{
  __shared__ __bf16 Vs[128 * 40];
  int pair = blockIdx.z, hh = blockIdx.y;
  int pBase = blockIdx.x * 128;
  int tid = threadIdx.x, lane = tid & 31, wave = tid >> 5;
  int g = lane >> 4;
  int vCh = (pair ? 768 : 128) + hh * 16;
  #pragma unroll
  for (int i = 0; i < 8; ++i) {
    int idx = tid + (i << 8);
    int d = idx >> 7, n = idx & 127;
    Vs[n * 40 + d]      = qkv[(size_t)(vCh + d) * HW + pBase + n];
    Vs[n * 40 + 16 + d] = (__bf16)0.0f;
  }
  __syncthreads();
  const __bf16* arow = attnB + ((size_t)(pair * 8 + hh) * 16 + (lane & 15)) * 32;
  v16bf a = cat8(*(const v8bf*)(arow + g * 8),
                 *(const v8bf*)(arow + 16 + g * 8));
  int n0 = wave * 16 + (lane & 15);
  v16bf b = cat8(*(const v8bf*)(Vs + n0 * 40 + g * 8),
                 *(const v8bf*)(Vs + n0 * 40 + 16 + g * 8));
  v8f acc = {};
  acc = wmma_bf(a, b, acc);
  int outCh = (pair ? 256 : 0) + hh * 16;
  int mg = (lane >> 4) * 8, n = lane & 15;
  #pragma unroll
  for (int jj = 0; jj < 8; ++jj) {
    int co = outCh + mg + jj;
    fusedIn[(size_t)co * HW + pBase + wave * 16 + n] = (__bf16)acc[jj];
  }
}

// ---------- per-pixel channel LayerNorm ----------
__global__ __launch_bounds__(256) void k_ln(
    const float* __restrict__ outF, const float* __restrict__ lw,
    const float* __restrict__ lb, __bf16* __restrict__ xnh)
{
  int p = blockIdx.x * 256 + threadIdx.x;
  float s = 0.f, s2 = 0.f;
  for (int c = 0; c < 128; ++c) {
    float v = outF[(size_t)c * HW + p];
    s += v; s2 += v * v;
  }
  float mu  = s * (1.f / 128.f);
  float var = s2 * (1.f / 128.f) - mu * mu;
  float inv = rsqrtf(var + 1e-5f);
  for (int c = 0; c < 128; ++c) {
    float v = (outF[(size_t)c * HW + p] - mu) * inv * lw[c] + lb[c];
    xnh[(size_t)c * HW + p] = (__bf16)v;
  }
}

// ---------- per-8x8-patch rfft2 * w -> irfft2 (separable 8-pt DFT) ----------
__constant__ float C8c[8] = {1.f, 0.70710678118654752f, 0.f, -0.70710678118654752f,
                             -1.f, -0.70710678118654752f, 0.f, 0.70710678118654752f};
__constant__ float S8c[8] = {0.f, 0.70710678118654752f, 1.f, 0.70710678118654752f,
                             0.f, -0.70710678118654752f, -1.f, -0.70710678118654752f};

__global__ __launch_bounds__(256) void k_fft(float* __restrict__ xp,
                                             const float* __restrict__ fw)
{
  int gid = blockIdx.x * 256 + threadIdx.x;   // 680*1024 threads
  int c = gid >> 10;
  int pat = gid & 1023;
  int py = pat >> 5, px = pat & 31;
  float* base = xp + (size_t)c * HW + (py << 3) * IMGW + (px << 3);
  float p[8][8];
  #pragma unroll
  for (int y = 0; y < 8; ++y)
    #pragma unroll
    for (int x = 0; x < 8; ++x) p[y][x] = base[y * IMGW + x];

  float Rre[8][5], Rim[8][5];
  #pragma unroll
  for (int y = 0; y < 8; ++y)
    #pragma unroll
    for (int v = 0; v < 5; ++v) {
      float sr = 0.f, si = 0.f;
      #pragma unroll
      for (int x = 0; x < 8; ++x) {
        int t = (v * x) & 7;
        sr += p[y][x] * C8c[t];
        si -= p[y][x] * S8c[t];
      }
      Rre[y][v] = sr; Rim[y][v] = si;
    }
  float Fre[8][5], Fim[8][5];
  #pragma unroll
  for (int u = 0; u < 8; ++u)
    #pragma unroll
    for (int v = 0; v < 5; ++v) {
      float ar = 0.f, ai = 0.f;
      #pragma unroll
      for (int y = 0; y < 8; ++y) {
        int t = (u * y) & 7;
        float cr = C8c[t], ci = -S8c[t];
        ar += Rre[y][v] * cr - Rim[y][v] * ci;
        ai += Rre[y][v] * ci + Rim[y][v] * cr;
      }
      float wv = fw[c * 40 + u * 5 + v];
      Fre[u][v] = ar * wv; Fim[u][v] = ai * wv;
    }
  float Gre[8][5], Gim[8][5];
  #pragma unroll
  for (int y = 0; y < 8; ++y)
    #pragma unroll
    for (int v = 0; v < 5; ++v) {
      float ar = 0.f, ai = 0.f;
      #pragma unroll
      for (int u = 0; u < 8; ++u) {
        int t = (u * y) & 7;
        float cr = C8c[t], ci = S8c[t];
        ar += Fre[u][v] * cr - Fim[u][v] * ci;
        ai += Fre[u][v] * ci + Fim[u][v] * cr;
      }
      Gre[y][v] = ar; Gim[y][v] = ai;
    }
  #pragma unroll
  for (int y = 0; y < 8; ++y)
    #pragma unroll
    for (int x = 0; x < 8; ++x) {
      float s = 0.f;
      #pragma unroll
      for (int v = 0; v < 8; ++v) {
        int vi = (v <= 4) ? v : 8 - v;
        float gr = Gre[y][vi];
        float gi = (v <= 4) ? Gim[y][vi] : -Gim[y][vi];
        int t = (v * x) & 7;
        s += gr * C8c[t] - gi * S8c[t];
      }
      base[y * IMGW + x] = s * (1.f / 64.f);
    }
}

// ---------- depthwise 3x3 + exact-GELU gate ----------
__global__ __launch_bounds__(256) void k_dwg(
    const float* __restrict__ xp, const float* __restrict__ dw,
    __bf16* __restrict__ xg)
{
  int gid = blockIdx.x * 256 + threadIdx.x;   // c*HW + p, c < 340
  int c = gid >> 16;
  int p = gid & 65535;
  int h = p >> 8, w = p & 255;
  const float* pa = xp + (size_t)c * HW;
  const float* pg = xp + (size_t)(c + 340) * HW;
  const float* wa = dw + c * 9;
  const float* wg = dw + (c + 340) * 9;
  float sa = 0.f, sg = 0.f;
  #pragma unroll
  for (int ky = 0; ky < 3; ++ky) {
    int hh = h + ky - 1;
    if (hh < 0 || hh > 255) continue;
    #pragma unroll
    for (int kx = 0; kx < 3; ++kx) {
      int ww = w + kx - 1;
      if (ww < 0 || ww > 255) continue;
      int off = hh * IMGW + ww;
      sa += wa[ky * 3 + kx] * pa[off];
      sg += wg[ky * 3 + kx] * pg[off];
    }
  }
  float ge = 0.5f * sa * (1.f + erff(sa * 0.70710678118654752f));
  xg[(size_t)c * HW + p] = (__bf16)(ge * sg);
}

extern "C" void kernel_launch(void* const* d_in, const int* in_sizes, int n_in,
                              void* d_out, int out_size, void* d_ws, size_t ws_size,
                              hipStream_t stream) {
  (void)in_sizes; (void)n_in; (void)out_size; (void)ws_size;
  const float* x   = (const float*)d_in[0];
  const float* c1w = (const float*)d_in[1];  const float* c1b = (const float*)d_in[2];
  const float* c2w = (const float*)d_in[3];  const float* c2b = (const float*)d_in[4];
  const float* c3w = (const float*)d_in[5];  const float* c3b = (const float*)d_in[6];
  const float* fuw = (const float*)d_in[7];  const float* fub = (const float*)d_in[8];
  const float* lnw = (const float*)d_in[9];  const float* lnb = (const float*)d_in[10];
  const float* t1  = (const float*)d_in[11]; const float* t2  = (const float*)d_in[12];
  const float* piw = (const float*)d_in[13];
  const float* dww = (const float*)d_in[14];
  const float* fftw= (const float*)d_in[15];
  const float* pow_= (const float*)d_in[16];

  char* ws = (char*)d_ws;
  const size_t HWs = 65536;
  // phase-1 region [0, ~218MB): xh + convOut + fusedIn
  __bf16* xh      = (__bf16*)(ws + 0);                    // 384 ch
  __bf16* convOut = (__bf16*)(ws + 50331648ULL);          // 896 ch: k1 v1 q1 q2 x2m k2 v2
  __bf16* fusedIn = (__bf16*)(ws + 167772160ULL);         // 384 ch: out1 x2m out2
  // phase-2 reuse (after fusion conv): xp f32 aliases xh/convOut/fusedIn
  float*  xp      = (float*)(ws + 0);                     // 680 ch f32
  __bf16* xg      = (__bf16*)(ws + 178257920ULL);         // 340 ch bf16
  float*  outF    = (float*)(ws + 222822400ULL);          // 128 ch f32 (fusion out / residual)
  __bf16* xnh     = (__bf16*)(ws + 256376832ULL);         // 128 ch bf16 (LN out)
  __bf16* A1   = (__bf16*)(ws + 273154048ULL);            // 256*1152
  __bf16* A2   = A1  + 294912;                            // 384*1152
  __bf16* A3   = A2  + 442368;                            // 256*1152
  __bf16* AF   = A3  + 294912;                            // 128*3456
  __bf16* API  = AF  + 442368;                            // 680*128
  __bf16* APO  = API + 87040;                             // 128*340
  __bf16* attnB= APO + 43520;                             // 2*8*16*32

  auto g1 = [](int n) { return dim3((unsigned)((n + 255) / 256)); };

  k_prep_w<<<g1(256*128*9), 256, 0, stream>>>(c1w, A1, 256, 128, 9);
  k_prep_w<<<g1(384*128*9), 256, 0, stream>>>(c2w, A2, 384, 128, 9);
  k_prep_w<<<g1(256*128*9), 256, 0, stream>>>(c3w, A3, 256, 128, 9);
  k_prep_w<<<g1(128*384*9), 256, 0, stream>>>(fuw, AF, 128, 384, 9);
  k_prep_w<<<g1(680*128),   256, 0, stream>>>(piw, API, 680, 128, 1);
  k_prep_w<<<g1(128*340),   256, 0, stream>>>(pow_, APO, 128, 340, 1);
  k_f2b<<<g1(384*65536), 256, 0, stream>>>(x, xh, 384*65536);

  // qkv convs
  k_gemm_conv<9><<<dim3(512, 4), 256, 0, stream>>>(xh,           A1, c1b, convOut,           nullptr, nullptr, 256, 128);
  k_gemm_conv<9><<<dim3(512, 6), 256, 0, stream>>>(xh + 128*HWs, A2, c2b, convOut + 256*HWs, nullptr, nullptr, 384, 128);
  k_gemm_conv<9><<<dim3(512, 4), 256, 0, stream>>>(xh + 256*HWs, A3, c3b, convOut + 640*HWs, nullptr, nullptr, 256, 128);
  // x2m slice -> fusion input middle
  hipMemcpyAsync(fusedIn + 128*HWs, convOut + 512*HWs, 128*HWs*sizeof(__bf16),
                 hipMemcpyDeviceToDevice, stream);
  // channel attention
  k_attn<<<16, 256, 0, stream>>>(convOut, t1, t2, attnB);
  k_av<<<dim3(512, 8, 2), 256, 0, stream>>>(convOut, attnB, fusedIn);
  // fusion conv -> f32 (residual + LN input)
  k_gemm_conv<9><<<dim3(512, 2), 256, 0, stream>>>(fusedIn, AF, fub, nullptr, outF, nullptr, 128, 384);
  // LayerNorm
  k_ln<<<256, 256, 0, stream>>>(outF, lnw, lnb, xnh);
  // project-in (1x1), f32 output for FFT filter
  k_gemm_conv<1><<<dim3(512, 11), 256, 0, stream>>>(xnh, API, nullptr, nullptr, xp, nullptr, 680, 128);
  // per-patch FFT filter (in place)
  k_fft<<<2720, 256, 0, stream>>>(xp, fftw);
  // depthwise conv + gelu gate
  k_dwg<<<87040, 256, 0, stream>>>(xp, dww, xg);
  // project-out (1x1) + residual -> d_out f32
  k_gemm_conv<1><<<dim3(512, 2), 256, 0, stream>>>(xg, APO, nullptr, nullptr, (float*)d_out, outF, 128, 340);
}
